// LSTMModel_27041114096277
// MI455X (gfx1250) — compile-verified, matrix operand order checked by reading, not run on hardware
//
#include <hip/hip_runtime.h>
#include <hip/hip_bf16.h>

typedef __attribute__((ext_vector_type(16))) _Float16 v16h;
typedef __attribute__((ext_vector_type(8)))  float    v8f;

#define WMMA_F32_F16(A, B, C) \
  __builtin_amdgcn_wmma_f32_16x16x32_f16(false, (A), false, (B), (short)0, (C), false, false)

__device__ __forceinline__ float sigmoidf_(float x) { return 1.f / (1.f + __expf(-x)); }

// ---- CDNA5 async global->LDS copy (16B per lane), tracked by ASYNCcnt ----
__device__ __forceinline__ void async_copy_b128(const void* g, const void* l) {
  // LDS aperture flat address: low 32 bits are the DS byte offset (ISA 10.2).
  asm volatile("global_load_async_to_lds_b128 %0, %1, off"
               :: "v"((unsigned)(uintptr_t)l), "v"(g)
               : "memory");
}

__device__ __forceinline__ void wait_async0() {
#if __has_builtin(__builtin_amdgcn_s_wait_asynccnt)
  __builtin_amdgcn_s_wait_asynccnt(0);
#else
  asm volatile("s_wait_asynccnt 0x0" ::: "memory");
#endif
}

// Stage the 16-row x_t tile (f32) into an LDS ping-pong buffer asynchronously.
template<int IN, int INP>
__device__ __forceinline__ void stage_x_async(const float* __restrict__ x, const float* dst,
                                              int bbase, int T, int t, int tid) {
  constexpr int CHR    = IN / 4;     // 16-byte chunks per row
  constexpr int CHUNKS = 16 * CHR;   // chunks per tile
  for (int ch = tid; ch < CHUNKS; ch += 256) {
    const int bm = ch / CHR;
    const int c4 = ch - bm * CHR;
    const float* g = x + ((size_t)(bbase + bm) * T + t) * IN + c4 * 4;
    const float* l = dst + bm * INP + c4 * 4;
    async_copy_b128(g, l);
  }
}

// A fragment (16x32 f16, MxK) built from row-major f32 LDS [16][LDK] at K-chunk k0.
// Lane m(0..15): halves 0..7 = K k0+koff..+7, halves 8..15 = K k0+16+koff..+23,
// koff = (lane>=16) ? 8 : 0.
template<int LDK>
__device__ __forceinline__ v16h load_a_frag_lds_f32(const float* base, int k0, int lane) {
  const int m    = lane & 15;
  const int koff = (lane >> 4) << 3;
  const float* p0 = base + m * LDK + k0 + koff;
  v16h a;
#pragma unroll
  for (int j = 0; j < 8; ++j) { a[j] = (_Float16)p0[j]; a[8 + j] = (_Float16)p0[16 + j]; }
  return a;
}

// Same, but from f16 LDS (used for h).
template<int LDK>
__device__ __forceinline__ v16h load_a_frag_lds(const _Float16* base, int k0, int lane) {
  const int m    = lane & 15;
  const int koff = (lane >> 4) << 3;
  const _Float16* p0 = base + m * LDK + k0 + koff;
  v16h a;
#pragma unroll
  for (int j = 0; j < 8; ++j) { a[j] = p0[j]; a[8 + j] = p0[16 + j]; }
  return a;
}

// B fragment (32x16 f16, KxN) for W^T, W row-major [G][KS] f32 in global.
// B[k][n] = W[nrow0+n][k]; lane n(0..15), koff=(lane>=16)?8:0. Bounds-checked so the
// K-chunk can be zero-padded (layer1: K=24 -> 32).
template<int KS>
__device__ __forceinline__ v16h load_b_frag_w(const float* __restrict__ W,
                                              int nrow0, int k0, int lane) {
  const int n    = lane & 15;
  const int koff = (lane >> 4) << 3;
  const float* p = W + (size_t)(nrow0 + n) * KS;
  v16h b;
#pragma unroll
  for (int j = 0; j < 8; ++j) {
    const int k1 = k0 + koff + j;
    const int k2 = k1 + 16;
    b[j]     = (k1 < KS) ? (_Float16)p[k1] : (_Float16)0.f;
    b[8 + j] = (k2 < KS) ? (_Float16)p[k2] : (_Float16)0.f;
  }
  return b;
}

// Fused LSTM layer: each workgroup owns 16 batch rows, iterates all T steps.
// Gates via v_wmma_f32_16x16x32_f16 with Wih^T / Whh^T resident in VGPRs.
// x tiles double-buffered in LDS via global_load_async_to_lds_b128 (ASYNCcnt),
// prefetching step t+1 while step t computes. Two barriers per step.
template<int IN, int INP, int H, bool LAST_ONLY>
__global__ __launch_bounds__(256) void lstm_layer_kernel(
    const float* __restrict__ x,    // [B][T][IN]
    const float* __restrict__ Wih,  // [4H][IN]
    const float* __restrict__ Whh,  // [4H][H]
    const float* __restrict__ bias, // [4H]
    float* __restrict__ y,          // [B][T][H], or [B][H] if LAST_ONLY
    int T) {
  constexpr int G   = 4 * H;
  constexpr int KCI = INP / 32;      // K-chunks for input proj
  constexpr int KCH = H / 32;        // K-chunks for recurrent proj
  constexpr int NW  = 8;             // waves per workgroup (256 threads, wave32)
  constexpr int TPW = (G / 16) / NW; // 16-wide N tiles per wave
  constexpr int RLD = G + 4;         // padded LDS row to break bank conflicts

  __shared__ float    sXf[2][16 * INP]; // ping-pong x_t tiles (f32, async-staged)
  __shared__ _Float16 sH[16 * H];       // h_{t-1} (f16)
  __shared__ float    sR[16 * RLD];     // gate pre-activations (WMMA result)
  __shared__ float    sBias[G];
  __shared__ float    sC[16 * H];       // cell state

  const int tid   = threadIdx.x;
  const int lane  = tid & 31;
  const int wave  = tid >> 5;
  const int bbase = blockIdx.x * 16;

  for (int i = tid; i < G; i += 256) sBias[i] = bias[i];
  for (int i = tid; i < 16 * H; i += 256) { sH[i] = (_Float16)0.f; sC[i] = 0.f; }
  if (IN != INP) {
    // zero the K-pad columns once; async copies never touch them
    for (int i = tid; i < 16 * INP; i += 256) {
      const int ic = i % INP;
      if (ic >= IN) { sXf[0][i] = 0.f; sXf[1][i] = 0.f; }
    }
  }

  // Resident weight fragments for this wave's gate-column tiles.
  v16h fih[TPW][KCI];
  v16h fhh[TPW][KCH];
#pragma unroll
  for (int tt = 0; tt < TPW; ++tt) {
    const int g0 = (wave * TPW + tt) * 16;
#pragma unroll
    for (int kc = 0; kc < KCI; ++kc) fih[tt][kc] = load_b_frag_w<IN>(Wih, g0, kc * 32, lane);
#pragma unroll
    for (int kc = 0; kc < KCH; ++kc) fhh[tt][kc] = load_b_frag_w<H>(Whh, g0, kc * 32, lane);
  }

  // Prime the pipeline: async-stage x_0 into buffer 0.
  stage_x_async<IN, INP>(x, sXf[0], bbase, T, 0, tid);
  wait_async0();
  __syncthreads();

  for (int t = 0; t < T; ++t) {
    const int cur = t & 1;

    // Prefetch x_{t+1} into the other buffer while we compute on this one.
    if (t + 1 < T) stage_x_async<IN, INP>(x, sXf[cur ^ 1], bbase, T, t + 1, tid);

    // Shared A fragments for this wave.
    v16h ax[KCI];
#pragma unroll
    for (int kc = 0; kc < KCI; ++kc) ax[kc] = load_a_frag_lds_f32<INP>(sXf[cur], kc * 32, lane);
    v16h ah[KCH];
#pragma unroll
    for (int kc = 0; kc < KCH; ++kc) ah[kc] = load_a_frag_lds<H>(sH, kc * 32, lane);

    // gates_pre = x_t * Wih^T + h_{t-1} * Whh^T  (one f32 accumulator chain)
#pragma unroll
    for (int tt = 0; tt < TPW; ++tt) {
      v8f acc = {};
#pragma unroll
      for (int kc = 0; kc < KCI; ++kc) acc = WMMA_F32_F16(ax[kc], fih[tt][kc], acc);
#pragma unroll
      for (int kc = 0; kc < KCH; ++kc) acc = WMMA_F32_F16(ah[kc], fhh[tt][kc], acc);
      const int g0 = (wave * TPW + tt) * 16;
      const int n  = lane & 15;
      const int mb = (lane >> 4) * 8;
#pragma unroll
      for (int vv = 0; vv < 8; ++vv) sR[(mb + vv) * RLD + g0 + n] = acc[vv];
    }
    __syncthreads();

    // Elementwise LSTM cell update (gate order i,f,g,o).
    for (int i = tid; i < 16 * H; i += 256) {
      const int bm = i / H, hc = i % H;
      const float* r = &sR[bm * RLD];
      const float gi = sigmoidf_(sBias[hc]         + r[hc]);
      const float gf = sigmoidf_(sBias[H + hc]     + r[H + hc]);
      const float gg = tanhf    (sBias[2 * H + hc] + r[2 * H + hc]);
      const float go = sigmoidf_(sBias[3 * H + hc] + r[3 * H + hc]);
      const float c  = gf * sC[i] + gi * gg;
      sC[i] = c;
      const float h = go * tanhf(c);
      sH[i] = (_Float16)h;
      if (!LAST_ONLY) {
        y[((size_t)(bbase + bm) * T + t) * H + hc] = h;
      } else if (t == T - 1) {
        y[(size_t)(bbase + bm) * H + hc] = h;
      }
    }

    // Retire this wave's async copies for t+1, then make them visible to all waves.
    wait_async0();
    __syncthreads();
  }
}

// Tiny MLP head: out[b] = relu(last[b] @ Wfc1^T + b1) @ Wfc2^T + b2
__global__ __launch_bounds__(256) void head_kernel(
    const float* __restrict__ last,  // [B][32]
    const float* __restrict__ Wfc1,  // [32][32]
    const float* __restrict__ bfc1,  // [32]
    const float* __restrict__ Wfc2,  // [1][32]
    const float* __restrict__ bfc2,  // [1]
    float* __restrict__ out, int Btot) {
  __shared__ float w1[32 * 32];
  __shared__ float b1[32];
  __shared__ float w2[32];
  for (int i = threadIdx.x; i < 32 * 32; i += 256) w1[i] = Wfc1[i];
  if (threadIdx.x < 32) {
    b1[threadIdx.x] = bfc1[threadIdx.x];
    w2[threadIdx.x] = Wfc2[threadIdx.x];
  }
  __syncthreads();
  const int b = blockIdx.x * 256 + threadIdx.x;
  if (b >= Btot) return;
  float xr[32];
#pragma unroll
  for (int k = 0; k < 32; ++k) xr[k] = last[(size_t)b * 32 + k];
  float acc = bfc2[0];
#pragma unroll
  for (int j = 0; j < 32; ++j) {
    float a = b1[j];
#pragma unroll
    for (int k = 0; k < 32; ++k) a += xr[k] * w1[j * 32 + k];
    acc += fmaxf(a, 0.f) * w2[j];
  }
  out[b] = acc;
}

extern "C" void kernel_launch(void* const* d_in, const int* in_sizes, int n_in,
                              void* d_out, int out_size, void* d_ws, size_t ws_size,
                              hipStream_t stream) {
  const float* x    = (const float*)d_in[0];
  const float* Wih1 = (const float*)d_in[1];
  const float* Whh1 = (const float*)d_in[2];
  const float* b1   = (const float*)d_in[3];
  const float* Wih2 = (const float*)d_in[4];
  const float* Whh2 = (const float*)d_in[5];
  const float* b2   = (const float*)d_in[6];
  const float* Wih3 = (const float*)d_in[7];
  const float* Whh3 = (const float*)d_in[8];
  const float* b3   = (const float*)d_in[9];
  const float* Wfc1 = (const float*)d_in[10];
  const float* bfc1 = (const float*)d_in[11];
  const float* Wfc2 = (const float*)d_in[12];
  const float* bfc2 = (const float*)d_in[13];
  float* out = (float*)d_out;

  const int B = 4096, T = 128;

  float* y1 = (float*)d_ws;                     // [B][T][128]
  float* y2 = y1 + (size_t)B * T * 128;         // [B][T][64]
  float* y3 = y2 + (size_t)B * T * 64;          // [B][32] (last step only)

  dim3 grid(B / 16), block(256);
  lstm_layer_kernel<24, 32, 128, false><<<grid, block, 0, stream>>>(x,  Wih1, Whh1, b1, y1, T);
  lstm_layer_kernel<128, 128, 64, false><<<grid, block, 0, stream>>>(y1, Wih2, Whh2, b2, y2, T);
  lstm_layer_kernel<64, 64, 32, true  ><<<grid, block, 0, stream>>>(y2, Wih3, Whh3, b3, y3, T);
  head_kernel<<<(B + 255) / 256, 256, 0, stream>>>(y3, Wfc1, bfc1, Wfc2, bfc2, out, B);
}
